// GNN_84885733638151
// MI455X (gfx1250) — compile-verified, matrix operand order checked by reading, not run on hardware
//
#include <hip/hip_runtime.h>
#include <hip/hip_bf16.h>

#define NN 100000
#define EE 640000
#define HH 128
#define CC 16

typedef __attribute__((ext_vector_type(16))) __bf16 v16bf;
typedef __attribute__((ext_vector_type(8)))  float  v8f;

// ---------------------------------------------------------------- zero
__global__ void zero_f32(float* __restrict__ p, int n) {
    int i = blockIdx.x * blockDim.x + threadIdx.x;
    if (i < n) p[i] = 0.0f;
}

// ---------------------------------------------------------------- weight packing
// Pack a K(=128) x ncol fp32 weight matrix into bf16 WMMA B fragments.
// Fragment f = tile*4*32 + kc*32 + lane holds 16 bf16 elements; element e maps to
//   K    = kc*32 + (e>>3)*16 + (lane>>4)*8 + (e&7)
//   Ncol = tile*16 + (lane&15)
// matching the 16-bit 32x16 B-matrix lane striping (ISA 7.12.2), so GEMM waves
// fetch each B fragment as one contiguous 32-byte per-lane vector load.
__global__ void pack_b_frags(const float* __restrict__ W, int ncol,
                             __bf16* __restrict__ out) {
    int idx = blockIdx.x * blockDim.x + threadIdx.x;
    int total = (ncol / 16) * 4 * 32;
    if (idx >= total) return;
    int lane = idx & 31;
    int kc   = (idx >> 5) & 3;
    int tile = idx >> 7;
    int n    = tile * 16 + (lane & 15);
    int half = lane >> 4;
    __bf16* o = out + (size_t)idx * 16;
#pragma unroll
    for (int e = 0; e < 16; ++e) {
        int k = kc * 32 + (e >> 3) * 16 + half * 8 + (e & 7);
        o[e] = (__bf16)W[k * ncol + n];
    }
}

// ---------------------------------------------------------------- scatter (mean-agg numerator)
// One thread per (edge, float4-chunk): gather from feat[src], atomically add
// into msgbuf[dst]. Chunk 0 also accumulates the in-degree (first layer only).
__global__ __launch_bounds__(256) void scatter_edges(
        const int* __restrict__ ei, const float* __restrict__ feat,
        float* __restrict__ msgbuf, float* __restrict__ deg, int addDeg) {
    int idx = blockIdx.x * blockDim.x + threadIdx.x;
    if (idx >= EE * (HH / 4)) return;
    int e = idx >> 5;           // HH/4 == 32 chunks per edge
    int c = (idx & 31) * 4;
    int src = ei[e];
    int dst = ei[EE + e];
    float* p = msgbuf + (size_t)dst * HH + c;
    __builtin_prefetch(p, 1, 0);                       // global_prefetch_b8 on dst row
    const float4 v = *(const float4*)(feat + (size_t)src * HH + c);
    atomicAdd(p + 0, v.x);
    atomicAdd(p + 1, v.y);
    atomicAdd(p + 2, v.z);
    atomicAdd(p + 3, v.w);
    if (addDeg && c == 0) atomicAdd(deg + dst, 1.0f);
}

// ---------------------------------------------------------------- layer-1 fused GEMM
// h = relu( (msg/deg) @ W1l + x @ W1r + b1 ), tile 16 rows/block, 8 waves cover
// the 128 output columns. A rows staged to LDS as bf16 and shared by all waves.
__global__ __launch_bounds__(256) void sage_gemm1(
        const float* __restrict__ msg, const float* __restrict__ deg,
        const float* __restrict__ x,
        const __bf16* __restrict__ Bl, const __bf16* __restrict__ Br,
        const float* __restrict__ bias, float* __restrict__ hout) {
    __shared__ __bf16 sAgg[16 * HH];
    __shared__ __bf16 sX[16 * HH];
    const int row0 = blockIdx.x * 16;
    const int t = threadIdx.x;

    // Stage 16 rows of agg (= msg/deg) and x as bf16.
    for (int i = t; i < 16 * (HH / 4); i += 256) {
        int r   = i >> 5;                 // HH/4 = 32 chunks per row
        int c   = (i & 31) * 4;
        int row = row0 + r;
        float rd  = 1.0f / fmaxf(deg[row], 1.0f);
        float4 mv = *(const float4*)(msg + (size_t)row * HH + c);
        float4 xv = *(const float4*)(x   + (size_t)row * HH + c);
        sAgg[r * HH + c + 0] = (__bf16)(mv.x * rd);
        sAgg[r * HH + c + 1] = (__bf16)(mv.y * rd);
        sAgg[r * HH + c + 2] = (__bf16)(mv.z * rd);
        sAgg[r * HH + c + 3] = (__bf16)(mv.w * rd);
        sX[r * HH + c + 0] = (__bf16)xv.x;
        sX[r * HH + c + 1] = (__bf16)xv.y;
        sX[r * HH + c + 2] = (__bf16)xv.z;
        sX[r * HH + c + 3] = (__bf16)xv.w;
    }
    __syncthreads();

    const int wave = t >> 5;              // col tile 0..7
    const int lane = t & 31;
    const int m    = lane & 15;
    const int half = lane >> 4;
    v8f acc = {};
#pragma unroll
    for (int kc = 0; kc < 4; ++kc) {
        const int kbase = kc * 32 + half * 8;
        const __bf16* pa = &sAgg[m * HH + kbase];
        const __bf16* px = &sX  [m * HH + kbase];
        v16bf aA, aX;
#pragma unroll
        for (int e = 0; e < 8; ++e) {     // A 16x32 bf16 lane striping (ISA 7.12.2)
            aA[e] = pa[e];  aA[e + 8] = pa[e + 16];
            aX[e] = px[e];  aX[e + 8] = px[e + 16];
        }
        const int f = (wave * 4 + kc) * 32 + lane;
        v16bf bA = *(const v16bf*)(Bl + (size_t)f * 16);
        v16bf bX = *(const v16bf*)(Br + (size_t)f * 16);
        acc = __builtin_amdgcn_wmma_f32_16x16x32_bf16(false, aA, false, bA,
                                                      (short)0, acc, false, false);
        acc = __builtin_amdgcn_wmma_f32_16x16x32_bf16(false, aX, false, bX,
                                                      (short)0, acc, false, false);
    }
    const int n = wave * 16 + (lane & 15);
    const float b = bias[n];
#pragma unroll
    for (int v = 0; v < 8; ++v) {
        int row = row0 + v + 8 * half;    // C/D: M = v + 8*(lane>>4), N = lane&15
        hout[(size_t)row * HH + n] = fmaxf(acc[v] + b, 0.0f);
    }
}

// ---------------------------------------------------------------- layer-2 fused GEMM
// out = (msg/deg) @ W2l + h @ W2r + b2   (C = 16 -> one col tile per wave)
__global__ __launch_bounds__(256) void sage_gemm2(
        const float* __restrict__ msg, const float* __restrict__ deg,
        const float* __restrict__ h,
        const __bf16* __restrict__ Bl, const __bf16* __restrict__ Br,
        const float* __restrict__ bias, float* __restrict__ out, int ntiles) {
    const int wave = threadIdx.x >> 5;
    const int lane = threadIdx.x & 31;
    const int tile = blockIdx.x * 8 + wave;
    if (tile >= ntiles) return;
    const int row0 = tile * 16;
    const int m    = lane & 15;
    const int half = lane >> 4;
    const int row  = row0 + m;
    const float rd = 1.0f / fmaxf(deg[row], 1.0f);
    const float* pm = msg + (size_t)row * HH;
    const float* ph = h   + (size_t)row * HH;

    v8f acc = {};
#pragma unroll
    for (int kc = 0; kc < 4; ++kc) {
        const int kbase = kc * 32 + half * 8;
        float4 m0 = *(const float4*)(pm + kbase);
        float4 m1 = *(const float4*)(pm + kbase + 4);
        float4 m2 = *(const float4*)(pm + kbase + 16);
        float4 m3 = *(const float4*)(pm + kbase + 20);
        float4 h0 = *(const float4*)(ph + kbase);
        float4 h1 = *(const float4*)(ph + kbase + 4);
        float4 h2 = *(const float4*)(ph + kbase + 16);
        float4 h3 = *(const float4*)(ph + kbase + 20);
        v16bf aA, aH;
        aA[0]=(__bf16)(m0.x*rd); aA[1]=(__bf16)(m0.y*rd); aA[2]=(__bf16)(m0.z*rd); aA[3]=(__bf16)(m0.w*rd);
        aA[4]=(__bf16)(m1.x*rd); aA[5]=(__bf16)(m1.y*rd); aA[6]=(__bf16)(m1.z*rd); aA[7]=(__bf16)(m1.w*rd);
        aA[8]=(__bf16)(m2.x*rd); aA[9]=(__bf16)(m2.y*rd); aA[10]=(__bf16)(m2.z*rd); aA[11]=(__bf16)(m2.w*rd);
        aA[12]=(__bf16)(m3.x*rd); aA[13]=(__bf16)(m3.y*rd); aA[14]=(__bf16)(m3.z*rd); aA[15]=(__bf16)(m3.w*rd);
        aH[0]=(__bf16)h0.x; aH[1]=(__bf16)h0.y; aH[2]=(__bf16)h0.z; aH[3]=(__bf16)h0.w;
        aH[4]=(__bf16)h1.x; aH[5]=(__bf16)h1.y; aH[6]=(__bf16)h1.z; aH[7]=(__bf16)h1.w;
        aH[8]=(__bf16)h2.x; aH[9]=(__bf16)h2.y; aH[10]=(__bf16)h2.z; aH[11]=(__bf16)h2.w;
        aH[12]=(__bf16)h3.x; aH[13]=(__bf16)h3.y; aH[14]=(__bf16)h3.z; aH[15]=(__bf16)h3.w;

        const int f = kc * 32 + lane;     // single col tile
        v16bf bA = *(const v16bf*)(Bl + (size_t)f * 16);
        v16bf bH = *(const v16bf*)(Br + (size_t)f * 16);
        acc = __builtin_amdgcn_wmma_f32_16x16x32_bf16(false, aA, false, bA,
                                                      (short)0, acc, false, false);
        acc = __builtin_amdgcn_wmma_f32_16x16x32_bf16(false, aH, false, bH,
                                                      (short)0, acc, false, false);
    }
    const int n = lane & 15;
    const float b = bias[n];
#pragma unroll
    for (int v = 0; v < 8; ++v)
        out[(size_t)(row0 + v + 8 * half) * CC + n] = acc[v] + b;
}

// ---------------------------------------------------------------- host launch
extern "C" void kernel_launch(void* const* d_in, const int* in_sizes, int n_in,
                              void* d_out, int out_size, void* d_ws, size_t ws_size,
                              hipStream_t stream) {
    const float* x   = (const float*)d_in[0];
    const int*   ei  = (const int*)d_in[1];
    const float* W1l = (const float*)d_in[2];
    const float* W1r = (const float*)d_in[3];
    const float* b1  = (const float*)d_in[4];
    const float* W2l = (const float*)d_in[5];
    const float* W2r = (const float*)d_in[6];
    const float* b2  = (const float*)d_in[7];
    float* out = (float*)d_out;

    // Workspace layout (~103 MB): deg | msg | h | packed bf16 weight frags
    char* ws = (char*)d_ws;
    const size_t NH = (size_t)NN * HH * sizeof(float);   // 51.2 MB
    float*  deg  = (float*)(ws);
    float*  msg  = (float*)(ws + 524288);
    float*  hbuf = (float*)(ws + 524288 + NH);
    __bf16* pW1l = (__bf16*)(ws + 524288 + 2 * NH);
    __bf16* pW1r = pW1l + 8 * 4 * 32 * 16;
    __bf16* pW2l = pW1r + 8 * 4 * 32 * 16;
    __bf16* pW2r = pW2l + 1 * 4 * 32 * 16;

    (void)in_sizes; (void)n_in; (void)out_size; (void)ws_size;

    zero_f32<<<(NN + 255) / 256, 256, 0, stream>>>(deg, NN);
    zero_f32<<<(NN * HH + 255) / 256, 256, 0, stream>>>(msg, NN * HH);

    pack_b_frags<<<4, 256, 0, stream>>>(W1l, HH, pW1l);
    pack_b_frags<<<4, 256, 0, stream>>>(W1r, HH, pW1r);
    pack_b_frags<<<1, 128, 0, stream>>>(W2l, CC, pW2l);
    pack_b_frags<<<1, 128, 0, stream>>>(W2r, CC, pW2r);

    // Layer 1
    scatter_edges<<<(EE * (HH / 4) + 255) / 256, 256, 0, stream>>>(ei, x, msg, deg, 1);
    sage_gemm1<<<NN / 16, 256, 0, stream>>>(msg, deg, x, pW1l, pW1r, b1, hbuf);

    // Layer 2
    zero_f32<<<(NN * HH + 255) / 256, 256, 0, stream>>>(msg, NN * HH);
    scatter_edges<<<(EE * (HH / 4) + 255) / 256, 256, 0, stream>>>(ei, hbuf, msg, deg, 0);
    sage_gemm2<<<(NN / 16 + 7) / 8, 256, 0, stream>>>(msg, deg, hbuf, pW2l, pW2r, b2,
                                                      out, NN / 16);
}
